// QConv2D_81724637708776
// MI455X (gfx1250) — matrix operand order repaired and checked by reading, not compiled, not run on hardware
//
#include <hip/hip_runtime.h>
#include <hip/hip_bf16.h>

typedef __attribute__((ext_vector_type(16))) _Float16 v16h;
typedef __attribute__((ext_vector_type(8)))  float    v8f;

#define I0 126
#define I1 126
#define HH 128
#define WW 128
#define CC 3
#define QQ 9
#define FF 2
#define LL 2
#define NPIX (32 * I0 * I1)         // 508032
#define WAVES_TOTAL (NPIX / 16)     // 31752
#define BLOCKS (WAVES_TOTAL / 8)    // 3969

// ---------------------------------------------------------------------------
// Kernel 1: build packed f16 A-matrix tiles (weights) in the WMMA A layout.
// Per f: D[m][n] = sum_k W[m][k] * Act[k][n], with
//   m = q*3 + kcomp  (27 rows, padded to 32 -> 2 m-tiles of 16)
//   k = 2*(c*9+q)+t  (54 cols, padded to 64 -> 2 k-chunks of 32; t=0 sin, t=1 cos)
//   W[m][k] = (q match) ? (t==0 ? Mrot[kcomp][0] : Mrot[kcomp][2]) : 0
// A tile k order used here (contiguous): lane m (h=lane>>4), element e -> K = e + 16h.
// Apack index: (((f*2+mtile)*2+chunk)*32 + lane)*16 + e
// ---------------------------------------------------------------------------
__global__ void qconv_pack_weights(const float* __restrict__ a_rz,
                                   const float* __restrict__ b_ry,
                                   _Float16* __restrict__ Apack) {
    int tid   = threadIdx.x;        // 0..255, one thread per (f,mtile,chunk,lane)
    int lane  = tid & 31;
    int chunk = (tid >> 5) & 1;
    int mtile = (tid >> 6) & 1;
    int f     = (tid >> 7) & 1;
    int h     = lane >> 4;
    int m     = (lane & 15) + mtile * 16;

    _Float16* dst = Apack + ((((f * 2 + mtile) * 2 + chunk) * 32) + lane) * 16;

    #pragma unroll
    for (int e = 0; e < 16; ++e) {
        int K   = e + 16 * h;
        int kk  = chunk * 32 + K;
        int idx = kk >> 1;
        int t   = kk & 1;
        float val = 0.f;
        if (idx < 27 && m < 27) {
            int c = idx / QQ, q = idx % QQ;
            int qp = m / 3, kcomp = m % 3;
            if (q == qp) {
                // Apply rotation chain to basis vectors e0 (sin col), e2 (cos col)
                float sx = 1.f, sy = 0.f, sz = 0.f;
                float cx = 0.f, cy = 0.f, cz = 1.f;
                for (int l = 0; l < LL; ++l) {
                    float aa = a_rz[((f * CC + c) * LL + l) * QQ + q];
                    float bb = b_ry[((f * CC + c) * LL + l) * QQ + q];
                    float ca = cosf(aa), sa = sinf(aa);
                    float cb = cosf(bb), sb = sinf(bb);
                    float nx, ny, nz;
                    nx = sx * ca - sy * sa; ny = sx * sa + sy * ca; sx = nx; sy = ny;
                    nx = sx * cb + sz * sb; nz = -sx * sb + sz * cb; sx = nx; sz = nz;
                    nx = cx * ca - cy * sa; ny = cx * sa + cy * ca; cx = nx; cy = ny;
                    nx = cx * cb + cz * sb; nz = -cx * sb + cz * cb; cx = nx; cz = nz;
                }
                float scol[3] = { sx, sy, sz };
                float ccol[3] = { cx, cy, cz };
                val = (t == 0) ? scol[kcomp] : ccol[kcomp];
            }
        }
        dst[e] = (_Float16)val;
    }
}

// ---------------------------------------------------------------------------
// Kernel 2: main. Wave = 16 pixels (N axis of WMMA). Lane pair (n, n+16)
// owns pixel n: h=0 computes activation idx {chunk*16+0..7}, h=1 idx
// {chunk*16+8..15}; padded slots stay zero. 8 wmma per wave (2 f x 2 mtile
// x 2 accumulating k-chunks). Results -> LDS -> coalesced b128 stores.
// ---------------------------------------------------------------------------
__global__ __launch_bounds__(256) void qconv_main(
        const float* __restrict__ x,
        const float* __restrict__ w_enc,
        const _Float16* __restrict__ Apack,
        float* __restrict__ out) {
    __shared__ __align__(16) float lds[8 * 16 * 54];   // 27648 B

    int lane = threadIdx.x & 31;
    int wib  = threadIdx.x >> 5;
    int h    = lane >> 4;
    int n    = lane & 15;
    int gwave = blockIdx.x * 8 + wib;
    int p = gwave * 16 + n;                 // pixel id, exact coverage

    int j  = p % I1;
    int t1 = p / I1;
    int i  = t1 % I0;
    int b  = t1 / I0;
    const float* xp = x + (((long)b * HH + i) * WW + j) * CC;

    // Prefetch the three input rows of this patch (global_prefetch_b8)
    __builtin_prefetch(xp, 0, 0);
    __builtin_prefetch(xp + (long)WW * CC, 0, 0);
    __builtin_prefetch(xp + 2L * WW * CC, 0, 0);

    // Activations in B-matrix layout: element e -> K = e + 16h; kk = chunk*32+K
    // pair e=2s,2s+1 -> idx = chunk*16 + 8h + s, t=0 sin / t=1 cos
    v16h Bv[2][2] = {};   // [f][chunk], zero padding slots
    #pragma unroll
    for (int chunk = 0; chunk < 2; ++chunk) {
        #pragma unroll
        for (int s = 0; s < 8; ++s) {
            int idx = chunk * 16 + h * 8 + s;
            if (idx < 27) {
                int c  = idx / QQ, q = idx % QQ;
                int di = q / 3,    dj = q % 3;
                float xv = xp[(di * WW + dj) * CC + c];
                #pragma unroll
                for (int f = 0; f < FF; ++f) {
                    float th = xv * w_enc[(f * CC + c) * QQ + q];
                    float sv = __sinf(th);
                    float cv = __cosf(th);
                    Bv[f][chunk][2 * s]     = (_Float16)sv;
                    Bv[f][chunk][2 * s + 1] = (_Float16)cv;
                }
            }
        }
    }

    float* wl = lds + wib * (16 * 54);
    #pragma unroll
    for (int f = 0; f < FF; ++f) {
        const v16h* Abase = ((const v16h*)Apack) + f * 128;  // f stride: 2*2*32 tiles
        v16h a00 = Abase[0 * 64 + 0 * 32 + lane];            // mtile0, chunk0
        v16h a01 = Abase[0 * 64 + 1 * 32 + lane];            // mtile0, chunk1
        v16h a10 = Abase[1 * 64 + 0 * 32 + lane];            // mtile1, chunk0
        v16h a11 = Abase[1 * 64 + 1 * 32 + lane];            // mtile1, chunk1

        v8f acc0 = {};
        acc0 = __builtin_amdgcn_wmma_f32_16x16x32_f16(false, a00, false, Bv[f][0],
                                                      (short)0, acc0, false, false);
        acc0 = __builtin_amdgcn_wmma_f32_16x16x32_f16(false, a01, false, Bv[f][1],
                                                      (short)0, acc0, false, false);
        v8f acc1 = {};
        acc1 = __builtin_amdgcn_wmma_f32_16x16x32_f16(false, a10, false, Bv[f][0],
                                                      (short)0, acc1, false, false);
        acc1 = __builtin_amdgcn_wmma_f32_16x16x32_f16(false, a11, false, Bv[f][1],
                                                      (short)0, acc1, false, false);

        // D layout: VGPR r -> M = r + 8h, column = pixel n. ReLU here.
        float* dst = wl + n * 54 + f * 27;
        #pragma unroll
        for (int r = 0; r < 8; ++r) {
            int m0 = r + 8 * h;          // 0..15
            dst[m0] = fmaxf(acc0[r], 0.f);
            int m1 = 16 + r + 8 * h;     // 16..31, keep only <27
            if (m1 < 27) dst[m1] = fmaxf(acc1[r], 0.f);
        }
    }

    // Wave-internal: DS ops are in-order per wave; wait + fence compiler.
    __asm__ volatile("s_wait_dscnt 0" ::: "memory");

    // Coalesced b128 stores of this wave's contiguous 3456-byte output region.
    const float* rl = lds + wib * (16 * 54);
    float* gdst = out + (long)gwave * (16 * 54);
    #pragma unroll
    for (int it = 0; it < 7; ++it) {
        int off = it * 128 + lane * 4;   // floats; region is 864 floats
        if (off < 864) {
            float4 v = *(const float4*)(rl + off);
            *(float4*)(gdst + off) = v;
        }
    }
}

extern "C" void kernel_launch(void* const* d_in, const int* in_sizes, int n_in,
                              void* d_out, int out_size, void* d_ws, size_t ws_size,
                              hipStream_t stream) {
    const float* x     = (const float*)d_in[0];   // (32,128,128,3)
    const float* w_enc = (const float*)d_in[1];   // (2,3,9)
    const float* a_rz  = (const float*)d_in[2];   // (2,3,2,9)
    const float* b_ry  = (const float*)d_in[3];   // (2,3,2,9)
    float* out = (float*)d_out;                   // (32,126,126,54)
    _Float16* Apack = (_Float16*)d_ws;            // 8192 B packed weight tiles

    qconv_pack_weights<<<1, 256, 0, stream>>>(a_rz, b_ry, Apack);
    qconv_main<<<BLOCKS, 256, 0, stream>>>(x, w_enc, Apack, out);
}
